// ScaledDotProductAttention_22050362097907
// MI455X (gfx1250) — matrix operand order repaired and checked by reading, not compiled, not run on hardware
//
#include <hip/hip_runtime.h>

// CDNA5 / gfx1250 flash-attention forward (causal), fp32 in/out.
// fp16 WMMA (f32 accumulate) + fp32 online softmax, 64-key tiles.
// Row sums on the matrix pipe (P @ ones); row max via v_permlane16_b32;
// raw v_exp_f32 / v_rcp_f32 for softmax math.

typedef __attribute__((ext_vector_type(16))) _Float16 v16h;
typedef __attribute__((ext_vector_type(8)))  _Float16 v8h;
typedef __attribute__((ext_vector_type(2)))  _Float16 v2h;
typedef __attribute__((ext_vector_type(8)))  float    v8f;

#define S_LEN  2048
#define DKC    64
#define DVC    64
#define QTILE  128     // query rows per block (8 waves x 16 rows)
#define KTILE  64      // keys per iteration (two 32-key groups)
#define WAVES  8
#define KP     72      // kTile pitch in halves (144B: 16B-aligned, conflict-free)
#define VP     72      // vTile pitch in halves (64 slots + pad)
#define PP     72      // P staging pitch in halves (64 slots + pad)

__device__ __forceinline__ float fast_exp2(float x) {
#if __has_builtin(__builtin_amdgcn_exp2f)
  return __builtin_amdgcn_exp2f(x);      // single v_exp_f32, FTZ is fine here
#else
  return exp2f(x);
#endif
}

__device__ __forceinline__ float fast_rcp(float x) {
#if __has_builtin(__builtin_amdgcn_rcpf)
  return __builtin_amdgcn_rcpf(x);
#else
  return 1.0f / x;
#endif
}

// 16-lane xor-butterfly max using v_permlane16_b32 (VALU, no LDS traffic).
__device__ __forceinline__ float rowmax16(float v) {
#if __has_builtin(__builtin_amdgcn_permlane16)
  v = fmaxf(v, __int_as_float(__builtin_amdgcn_permlane16(
          __float_as_int(v), __float_as_int(v), 0x67452301u, 0xEFCDAB89u, false, false))); // xor 1
  v = fmaxf(v, __int_as_float(__builtin_amdgcn_permlane16(
          __float_as_int(v), __float_as_int(v), 0x54761032u, 0xDCFE98BAu, false, false))); // xor 2
  v = fmaxf(v, __int_as_float(__builtin_amdgcn_permlane16(
          __float_as_int(v), __float_as_int(v), 0x32107654u, 0xBA98FEDCu, false, false))); // xor 4
  v = fmaxf(v, __int_as_float(__builtin_amdgcn_permlane16(
          __float_as_int(v), __float_as_int(v), 0xFEDCBA98u, 0x76543210u, false, false))); // xor 8
#else
#pragma unroll
  for (int m = 1; m < 16; m <<= 1) v = fmaxf(v, __shfl_xor(v, m, 32));
#endif
  return v;
}

// Contraction-slot permutation (within a 32-key group) for the interleaved
// P staging layout. P A-fragment element j (lane-half hi) holds local key
// (hi*8 + (j>>1) + 16*(j&1)); V rows scatter to slot(key) so A and B agree.
__device__ __forceinline__ int keyslot(int k) {
  const int a = k >> 4, b = k & 15;
  const int inner = b;
  return (inner << 1) | a;
}

// Cooperative fill of one 64-key tile: K (row-major f16) and V (transposed,
// key-permuted f16). 256 threads: thread = key-row tid/4, features (tid%4)*16..+15.
__device__ __forceinline__ void fill_tiles(const float* __restrict__ Kb,
                                           const float* __restrict__ Vb,
                                           _Float16* __restrict__ kT,
                                           _Float16* __restrict__ vT,
                                           int kb, int tid) {
  const int row = tid >> 2;          // 0..63 (key within tile)
  const int col = (tid & 3) * 16;    // 0,16,32,48 (feature)
  {
    const float* src = Kb + (size_t)(kb + row) * DKC + col;
    v8h a, b;
#pragma unroll
    for (int e = 0; e < 8; ++e) { a[e] = (_Float16)src[e]; b[e] = (_Float16)src[8 + e]; }
    *(v8h*)(kT + row * KP + col)     = a;         // ds_store_b128
    *(v8h*)(kT + row * KP + col + 8) = b;
  }
  {
    const float* src = Vb + (size_t)(kb + row) * DVC + col;
    const int slot = (row & 32) | keyslot(row & 31);
#pragma unroll
    for (int j = 0; j < 16; ++j)                  // transposed scatter (b16)
      vT[(col + j) * VP + slot] = (_Float16)src[j];
  }
}

__global__ __launch_bounds__(256, 2)
void fa_fwd_causal_gfx1250(const float* __restrict__ Q,
                           const float* __restrict__ K,
                           const float* __restrict__ V,
                           float* __restrict__ O) {
  __shared__ __align__(16) _Float16 kTile[2][64 * KP];   // 18.4 KB
  __shared__ __align__(16) _Float16 vTile[2][64 * VP];   // 18.4 KB
  __shared__ __align__(16) _Float16 ldsP[WAVES][16 * PP]; // 18.4 KB

  const int tid  = threadIdx.x;
  const int lane = tid & 31;
  const int wave = tid >> 5;
  const int hi   = lane >> 4;     // half-wave select (wave32 fragment layout)
  const int ln   = lane & 15;

  const int qblocks = S_LEN / QTILE;          // 16
  const int bh   = blockIdx.x / qblocks;      // 0..31 (b*H + h)
  const int qblk = blockIdx.x % qblocks;
  const int q0   = qblk * QTILE + wave * 16;  // first query row of this wave

  const float qscale = 0.125f * 1.44269504088896f; // 1/sqrt(DK) * log2(e)

  // ---- Q A-fragments (16x32 f16), two contraction chunks over DK=64 ----
  const float* qrow = Q + ((size_t)bh * S_LEN + (size_t)(q0 + ln)) * DKC;
  v16h qa[2];
#pragma unroll
  for (int c = 0; c < 2; ++c) {
#pragma unroll
    for (int e = 0; e < 8; ++e) {
      qa[c][e]     = (_Float16)(qrow[c * 32 + hi * 16 + e]      * qscale);
      qa[c][e + 8] = (_Float16)(qrow[c * 32 + hi * 16 + 8 + e] * qscale);
    }
  }

  // All-ones B fragment: P @ ones -> row sums in every column of o[4].
  v16h ones;
#pragma unroll
  for (int e = 0; e < 16; ++e) ones[e] = (_Float16)1.0f;

  // ---- running softmax state: o[0..3] = P@V, o[4] = row-sum fragment ----
  float mrow[8];
  v8f o[5];
#pragma unroll
  for (int r = 0; r < 8; ++r) mrow[r] = -1e30f;
#pragma unroll
  for (int g = 0; g < 5; ++g) {
#pragma unroll
    for (int r = 0; r < 8; ++r) o[g][r] = 0.0f;
  }

  const float* Kb = K + (size_t)bh * S_LEN * DKC;
  const float* Vb = V + (size_t)bh * S_LEN * DVC;
  _Float16* myP = &ldsP[wave][0];

  const int block_nt = (qblk * QTILE + QTILE) >> 6;   // 64-key tiles, uniform

  fill_tiles(Kb, Vb, kTile[0], vTile[0], 0, tid);

  for (int t = 0; t < block_nt; ++t) {
    __syncthreads();                     // fill(t) visible to all waves
    const int kb = t << 6;

    if (t + 1 < block_nt) {
      fill_tiles(Kb, Vb, kTile[(t + 1) & 1], vTile[(t + 1) & 1], kb + KTILE, tid);
      if (t + 2 < block_nt)              // gfx1250 global_prefetch_b8
        __builtin_prefetch(Kb + (size_t)(kb + 2 * KTILE + (tid >> 2)) * DKC, 0, 1);
    }

    if (kb < q0 + 16) {                  // skip tiles fully above the diagonal
      const _Float16* kt = kTile[t & 1];
      const _Float16* vt = vTile[t & 1];

      // ---- scores: four 16x16 fragments (64 keys) ----
      v8f s[4];
#pragma unroll
      for (int f = 0; f < 4; ++f) {
        v16h k0 = *(const v16h*)(kt + (f * 16 + ln) * KP + hi * 16);
        v16h k1 = *(const v16h*)(kt + (f * 16 + ln) * KP + 32 + hi * 16);
        v8f z = {};
        z = __builtin_amdgcn_wmma_f32_16x16x32_f16(false, qa[0], false, k0, (short)0, z, false, false);
        s[f] = __builtin_amdgcn_wmma_f32_16x16x32_f16(false, qa[1], false, k1, (short)0, z, false, false);
      }

      // ---- causal mask + online softmax (log2 domain) ----
      const bool needMask = (kb + KTILE - 1 > q0);
#pragma unroll
      for (int r = 0; r < 8; ++r) {
        const int qr = q0 + r + hi * 8;      // C layout: row = r + hi*8
        if (needMask) {
#pragma unroll
          for (int f = 0; f < 4; ++f)
            if (kb + f * 16 + ln > qr) s[f][r] = -1e30f;
        }
        const float tmax = rowmax16(fmaxf(fmaxf(s[0][r], s[1][r]),
                                          fmaxf(s[2][r], s[3][r])));
        const float mnew = fmaxf(mrow[r], tmax);
        const float corr = fast_exp2(mrow[r] - mnew);   // v_exp_f32
        const float p0   = fast_exp2(s[0][r] - mnew);
        const float p1   = fast_exp2(s[1][r] - mnew);
        const float p2   = fast_exp2(s[2][r] - mnew);
        const float p3   = fast_exp2(s[3][r] - mnew);
        mrow[r] = mnew;
#pragma unroll
        for (int g = 0; g < 5; ++g) o[g][r] *= corr;    // incl. row-sum fragment
        // stage P interleaved by key: two packed b32 stores per row
        v2h pk01; pk01[0] = (_Float16)p0; pk01[1] = (_Float16)p1;
        v2h pk23; pk23[0] = (_Float16)p2; pk23[1] = (_Float16)p3;
        const int prow = (r + hi * 8) * PP;
        *(v2h*)(myP + prow + ln * 2)      = pk01;
        *(v2h*)(myP + prow + 32 + ln * 2) = pk23;
      }

      // ---- reload P as two A-fragments (permuted contraction order) ----
      const v16h pa0 = *(const v16h*)(myP + ln * PP + hi * 16);
      const v16h pa1 = *(const v16h*)(myP + ln * PP + 32 + hi * 16);

      // ---- O += P @ V (key-permuted V tile), plus row-sum fragment ----
#pragma unroll
      for (int g = 0; g < 4; ++g) {
        const v16h vf0 = *(const v16h*)(vt + (g * 16 + ln) * VP + hi * 16);
        o[g] = __builtin_amdgcn_wmma_f32_16x16x32_f16(false, pa0, false, vf0, (short)0, o[g], false, false);
      }
#pragma unroll
      for (int g = 0; g < 4; ++g) {
        const v16h vf1 = *(const v16h*)(vt + (g * 16 + ln) * VP + 32 + hi * 16);
        o[g] = __builtin_amdgcn_wmma_f32_16x16x32_f16(false, pa1, false, vf1, (short)0, o[g], false, false);
      }
      o[4] = __builtin_amdgcn_wmma_f32_16x16x32_f16(false, pa0, false, ones, (short)0, o[4], false, false);
      o[4] = __builtin_amdgcn_wmma_f32_16x16x32_f16(false, pa1, false, ones, (short)0, o[4], false, false);
    }
  }

  // ---- epilogue: normalize by the row-sum fragment (v_rcp_f32), store fp32 ----
  float* Ob = O + (size_t)bh * S_LEN * DVC;
#pragma unroll
  for (int r = 0; r < 8; ++r) {
    const float inv = fast_rcp(o[4][r]);
    const int row = q0 + r + hi * 8;
#pragma unroll
    for (int g = 0; g < 4; ++g)
      Ob[(size_t)row * DVC + g * 16 + ln] = o[g][r] * inv;
  }
}

extern "C" void kernel_launch(void* const* d_in, const int* in_sizes, int n_in,
                              void* d_out, int out_size, void* d_ws, size_t ws_size,
                              hipStream_t stream) {
  (void)in_sizes; (void)n_in; (void)out_size; (void)d_ws; (void)ws_size;
  const float* Q = (const float*)d_in[0];
  const float* K = (const float*)d_in[1];
  const float* V = (const float*)d_in[2];
  // d_in[3] is the causal mask; it is implemented analytically in-kernel.
  float* O = (float*)d_out;

  const int BH = 2 * 16;                 // B*H
  dim3 grid(BH * (S_LEN / QTILE));       // 512 blocks
  dim3 block(256);                       // 8 waves (wave32)
  fa_fwd_causal_gfx1250<<<grid, block, 0, stream>>>(Q, K, V, O);
}
